// LSTMAutoRegressor_37847251812902
// MI455X (gfx1250) — matrix operand order repaired and checked by reading, not compiled, not run on hardware
//
#include <hip/hip_runtime.h>
#include <hip/hip_bf16.h>

typedef __attribute__((ext_vector_type(16))) __bf16 v16bf;
typedef __attribute__((ext_vector_type(8)))  float  v8f;
typedef __attribute__((ext_vector_type(4)))  float  v4f;

#define B_TOT     4096
#define T_STEPS   64
#define F_DIM     64
#define U_DIM     256
#define GATE4     1024      // 4*U
#define KTOT      320       // F + U
#define NKC       10        // KTOT / 32
#define NTG       64        // GATE4 / 16
#define OUTD      65
#define OUT_STEPS 24
#define BTILE     32
#define NTHREADS  512       // 16 waves

static __device__ __forceinline__ unsigned short f2bf(float f) {
    union { float f; unsigned u; } v; v.f = f;
    unsigned r = v.u + 0x7FFFu + ((v.u >> 16) & 1u);   // round-to-nearest-even
    return (unsigned short)(r >> 16);
}

static __device__ __forceinline__ float sigmoidf(float x) {
    return 1.0f / (1.0f + __expf(-x));
}

// Load a 32-byte B fragment at a compile-time element offset from a per-lane
// base, through an explicit global (addrspace 1) pointer: lowers to
// global_load_b128 vdst, v[base], off offset:<imm>  (one shared address pair).
static __device__ __forceinline__ v16bf load_wfrag(const unsigned short* base, int eoff) {
    const __attribute__((address_space(1))) unsigned short* gp =
        (const __attribute__((address_space(1))) unsigned short*)(unsigned long long)base;
    return *(const __attribute__((address_space(1))) v16bf*)(gp + eoff);
}

// ---------------------------------------------------------------------------
// Prep 1: pack W = [K ; R] (320 x 1024 f32) into bf16 WMMA B-fragment blocks:
// frag(kc, ntg) = 32 lanes x 16 bf16, lane l holds n = ntg*16 + (l&15),
// k = kc*32 + (e>>3)*16 + (l>>4)*8 + (e&7).
// ---------------------------------------------------------------------------
__global__ void wb_prep_kernel(const float* __restrict__ Km,
                               const float* __restrict__ Rm,
                               unsigned short* __restrict__ Wb) {
    int flat = blockIdx.x * blockDim.x + threadIdx.x;
    if (flat >= KTOT * GATE4) return;
    int e    = flat & 15;
    int lane = (flat >> 4) & 31;
    int ntg  = (flat >> 9) & 63;
    int kc   = flat >> 15;
    int n = ntg * 16 + (lane & 15);
    int k = kc * 32 + ((e >> 3) << 4) + (lane >> 4) * 8 + (e & 7);
    float v = (k < F_DIM) ? Km[k * GATE4 + n] : Rm[(k - F_DIM) * GATE4 + n];
    Wb[flat] = f2bf(v);
}

// Prep 2: transpose dense_w (256 x 65) -> dwT (65 x 256) for contiguous loads.
__global__ void dwt_prep_kernel(const float* __restrict__ dw,
                                float* __restrict__ dwT) {
    int flat = blockIdx.x * blockDim.x + threadIdx.x;
    if (flat >= OUTD * U_DIM) return;
    int c = flat >> 8;          // 0..64
    int u = flat & 255;         // 0..255
    dwT[c * U_DIM + u] = dw[u * OUTD + c];
}

// ---------------------------------------------------------------------------
// Persistent LSTM-AR kernel: one WG (512 thr / 16 waves) owns BTILE=32 batch
// rows through 64 warm + 23 AR steps. Wave (mh,q): 16 batch rows x 32 u-cols
// across all 4 gates -> 8 accumulator tiles, 80 WMMAs per wave per step.
// Weights stay L2-resident (640 KB bf16); fragments are re-streamed each step
// with one k-chunk of software pipelining to hide L2 latency behind WMMA.
// ---------------------------------------------------------------------------
__global__ __launch_bounds__(NTHREADS, 1)
void lstm_ar_kernel(const float* __restrict__ inputs,
                    const float* __restrict__ bias,
                    const float* __restrict__ dense_b,
                    const float* __restrict__ dwT,
                    const unsigned short* __restrict__ Wb,
                    float* __restrict__ out) {
    // A-fragments of [x|h] in hardware A layout (bf16), per 16-row half-tile.
    __shared__ __align__(32) unsigned short afrag[2][NKC][32][16];   // 20 KB
    __shared__ float lds_hf[BTILE][U_DIM];                           // 32 KB
    __shared__ float lds_pred[BTILE][OUTD];                          // ~8 KB

    const int tid  = threadIdx.x;
    const int lane = tid & 31;
    const int wv   = tid >> 5;        // 0..15
    const int mh   = wv >> 3;         // 0..1 : which 16-row half of the tile
    const int q    = wv & 7;          // 0..7 : which 32-wide u-column slice
    const int lrow = lane >> 4;
    const int lcol = lane & 15;
    const int b0   = blockIdx.x * BTILE;

    // zero A-fragments (h0 = 0)
    {
        unsigned short* p = &afrag[0][0][0][0];
        for (int i = tid; i < 2 * NKC * 32 * 16; i += NTHREADS) p[i] = 0;
    }

    // per-lane bias for each of 8 accumulator n-subtiles (nt = gate*2 + j)
    float biasr[8];
    #pragma unroll
    for (int g = 0; g < 4; ++g)
        #pragma unroll
        for (int j = 0; j < 2; ++j)
            biasr[g * 2 + j] = bias[g * U_DIM + q * 32 + j * 16 + lcol];

    v8f cstate[2];
    #pragma unroll
    for (int j = 0; j < 2; ++j)
        #pragma unroll
        for (int r = 0; r < 8; ++r) cstate[j][r] = 0.0f;

    // x-fragment writer coordinates: thread -> (batch row, 4 consecutive k)
    const int xm  = tid >> 4;          // 0..31
    const int xk0 = (tid & 15) * 4;    // 0..60
    const int xmh = xm >> 4;
    const int xml = xm & 15;
    const float* xrow = inputs + (size_t)(b0 + xm) * T_STEPS * F_DIM + xk0;

    // Per-lane fragment base with the runtime (lane, q) terms folded in once:
    // fragment (kc,g,j) then sits at the COMPILE-TIME element offset
    // ((kc*64 + g*16 + j) << 9), i.e. a pure immediate on every load.
    const unsigned short* Wbase = Wb + lane * 16 + (q * 2 << 9);

    v8f hreg[2];

    for (int t = 0; t < T_STEPS + OUT_STEPS - 1; ++t) {
        // ---- stage x into A-fragments (k-chunks 0..1) ----
        if (t < T_STEPS) {
            v4f x0 = *(const v4f*)(xrow + (size_t)t * F_DIM);
            #pragma unroll
            for (int i = 0; i < 4; ++i) {
                int k  = xk0 + i;
                int kc = k >> 5, kl = k & 31;
                int fh = (kl >> 3) & 1;
                int fe = kl - ((kl >> 4) << 3) - (fh << 3);
                afrag[xmh][kc][fh * 16 + xml][fe] = f2bf(x0[i]);
            }
        } else {
            #pragma unroll
            for (int i = 0; i < 4; ++i) {
                int k  = xk0 + i;
                int kc = k >> 5, kl = k & 31;
                int fh = (kl >> 3) & 1;
                int fe = kl - ((kl >> 4) << 3) - (fh << 3);
                afrag[xmh][kc][fh * 16 + xml][fe] = f2bf(lds_pred[xm][k]);
            }
        }
        __syncthreads();   // fragments ready for all waves

        // ---- GEMM: z = [x|h] @ W + bias via WMMA bf16, 1-chunk pipelined ----
        v8f acc[8];
        #pragma unroll
        for (int nt = 0; nt < 8; ++nt)
            #pragma unroll
            for (int r = 0; r < 8; ++r) acc[nt][r] = biasr[nt];

        const unsigned short* Wl = Wbase;
        asm volatile("" : "+v"(Wl) : :);   // fresh, non-invariant base each step

        v16bf acur, anxt;
        v16bf bcur[8], bnxt[8];

        // prologue: chunk 0 fragments (immediate offsets from the single base)
        acur = *reinterpret_cast<const v16bf*>(&afrag[mh][0][lane][0]);
        #pragma unroll
        for (int g = 0; g < 4; ++g)
            #pragma unroll
            for (int j = 0; j < 2; ++j)
                bcur[g * 2 + j] = load_wfrag(Wl, (g * 16 + j) << 9);

        #pragma unroll
        for (int kc = 0; kc < NKC; ++kc) {
            // pin this pipeline stage: nothing crosses chunk boundaries
            asm volatile("" : "+v"(Wl) : : "memory");
            if (kc + 1 < NKC) {
                anxt = *reinterpret_cast<const v16bf*>(&afrag[mh][kc + 1][lane][0]);
                #pragma unroll
                for (int g = 0; g < 4; ++g)
                    #pragma unroll
                    for (int j = 0; j < 2; ++j)
                        bnxt[g * 2 + j] = load_wfrag(
                            Wl, ((kc + 1) * NTG + g * 16 + j) << 9);
            }
            #pragma unroll
            for (int g = 0; g < 4; ++g)
                #pragma unroll
                for (int j = 0; j < 2; ++j)
                    acc[g * 2 + j] = __builtin_amdgcn_wmma_f32_16x16x32_bf16(
                        false, acur, false, bcur[g * 2 + j],
                        (short)0, acc[g * 2 + j], false, false);
            acur = anxt;
            #pragma unroll
            for (int nt = 0; nt < 8; ++nt) bcur[nt] = bnxt[nt];
        }

        // ---- elementwise LSTM cell (lane-local: all 4 gates in this wave) ----
        #pragma unroll
        for (int j = 0; j < 2; ++j) {
            #pragma unroll
            for (int r = 0; r < 8; ++r) {
                float iv = sigmoidf(acc[0 + j][r]);
                float fv = sigmoidf(acc[2 + j][r]);
                float gv = tanhf   (acc[4 + j][r]);
                float ov = sigmoidf(acc[6 + j][r]);
                float c  = fv * cstate[j][r] + iv * gv;
                cstate[j][r] = c;
                hreg[j][r]   = ov * tanhf(c);
            }
        }
        __syncthreads();   // all WMMA reads of afrag complete before rewrite

        // ---- write h back: bf16 A-fragments (k-chunks 2..9) + f32 copy ----
        #pragma unroll
        for (int j = 0; j < 2; ++j) {
            #pragma unroll
            for (int r = 0; r < 8; ++r) {
                int ml = r + 8 * lrow;
                int u  = q * 32 + j * 16 + lcol;
                int kk = 64 + u;
                int kc = kk >> 5, kl = kk & 31;
                int fh = (kl >> 3) & 1;
                int fe = kl - ((kl >> 4) << 3) - (fh << 3);
                float h = hreg[j][r];
                afrag[mh][kc][fh * 16 + ml][fe] = f2bf(h);
                lds_hf[mh * 16 + ml][u] = h;
            }
        }

        // ---- dense head: pred = h @ dense_w + dense_b (warm end + AR steps) ----
        if (t >= T_STEPS - 1) {
            __syncthreads();   // lds_hf complete
            int step = t - (T_STEPS - 1);
            int m  = tid >> 4;       // 0..31
            int c0 = tid & 15;
            for (int c = c0; c < OUTD; c += 16) {
                const float* wrow = dwT + c * U_DIM;   // contiguous in u
                float s = dense_b[c];
                for (int u = 0; u < U_DIM; u += 4) {
                    v4f h4 = *(const v4f*)&lds_hf[m][u];
                    v4f w4 = *(const v4f*)(wrow + u);
                    s += h4[0]*w4[0] + h4[1]*w4[1] + h4[2]*w4[2] + h4[3]*w4[3];
                }
                lds_pred[m][c] = s;
                if (c == OUTD - 1)
                    out[(size_t)(b0 + m) * OUT_STEPS + step] = s;
            }
            __syncthreads();   // pred ready to feed back as next x
        }
    }
}

// ---------------------------------------------------------------------------
extern "C" void kernel_launch(void* const* d_in, const int* in_sizes, int n_in,
                              void* d_out, int out_size, void* d_ws, size_t ws_size,
                              hipStream_t stream) {
    const float* inputs = (const float*)d_in[0];
    const float* Km     = (const float*)d_in[1];
    const float* Rm     = (const float*)d_in[2];
    const float* bias   = (const float*)d_in[3];
    const float* dw     = (const float*)d_in[4];
    const float* db     = (const float*)d_in[5];
    unsigned short* Wb  = (unsigned short*)d_ws;               // 640 KB bf16
    float* dwT          = (float*)((char*)d_ws + (size_t)KTOT * GATE4 * 2);
    float* out          = (float*)d_out;

    int prep_elems = KTOT * GATE4;
    wb_prep_kernel<<<(prep_elems + 255) / 256, 256, 0, stream>>>(Km, Rm, Wb);
    int dwt_elems = OUTD * U_DIM;
    dwt_prep_kernel<<<(dwt_elems + 255) / 256, 256, 0, stream>>>(dw, dwT);
    lstm_ar_kernel<<<B_TOT / BTILE, NTHREADS, 0, stream>>>(inputs, bias, db, dwT, Wb, out);
}
